// VectorQuantizer_3169685864681
// MI455X (gfx1250) — matrix operand order repaired and checked by reading, not compiled, not run on hardware
//
#include <hip/hip_runtime.h>
#include <hip/hip_bf16.h>
#include <math.h>

// ---------------------------------------------------------------------------
// VQ-VAE vector quantizer, MI455X (gfx1250, wave32, WMMA).
//   N = 16384 input vectors, D = 64, K = 8192 codewords.
//   argmin_k ( ||w_k||^2 - 2 x.w_k )   [ ||x||^2 is row-constant -> dropped ]
//   GEMM done in f16 WMMA 16x16x32 with f32 accumulate; codebook pre-scaled
//   by 2^13 (exact power of two -> argmin-invariant) so its
//   uniform(-1/8192,1/8192) values sit in f16's normal range; scores rescaled
//   by 2^-12 (= 2 * 2^-13).  B fragments are staged block-wide through LDS
//   (8 waves share each column tile) to cut L2 traffic 8x.
// ---------------------------------------------------------------------------

typedef __attribute__((ext_vector_type(16))) _Float16 v16h;
typedef __attribute__((ext_vector_type(8)))  float    v8f;

#define VQ_N      16384
#define VQ_D      64
#define VQ_K      8192
#define VQ_NELEM  (VQ_N * VQ_D)        // 1048576
#define ROW_TILES (VQ_N / 16)          // 1024
#define COL_TILES (VQ_K / 16)          // 512
#define TILES_PER_PHASE 16             // 16 col tiles * 2KB = 32KB LDS
#define NUM_PHASES (COL_TILES / TILES_PER_PHASE) // 32
#define GATHER_BLOCKS (VQ_NELEM / 256) // 4096

// ---- pack inputs into WMMA A-fragment order (16-bit A 16x32 layout) -------
// frag (mt, c): 512 halves, half index = lane*16 + j
//   lane<16 : row = mt*16+lane,     k = c*32 + (j<8 ? j   : j+8)
//   lane>=16: row = mt*16+lane-16,  k = c*32 + (j<8 ? j+8 : j+16)
__global__ __launch_bounds__(256) void vq_pack_x(const float* __restrict__ x,
                                                 _Float16* __restrict__ xpack) {
  int t    = blockIdx.x * 256 + threadIdx.x;   // one thread per packed half
  int frag = t >> 9;
  int r    = t & 511;
  int lane = r >> 4;
  int j    = r & 15;
  int mt   = frag >> 1;
  int c    = frag & 1;
  int row  = mt * 16 + (lane & 15);
  int k    = (lane < 16) ? ((j < 8) ? j : j + 8)
                         : ((j < 8) ? j + 8 : j + 16);
  k += c * 32;
  xpack[t] = (_Float16)x[row * VQ_D + k];
}

// ---- pack codebook*2^13 into WMMA B-fragment order (16-bit B 32x16) -------
// frag (nt, c): lane<16: col = nt*16+lane, k = c*32 + (j<8 ? j : j+8)
//               lane>=16: col = nt*16+lane-16, k = c*32 + (j<8 ? j+8 : j+16)
__global__ __launch_bounds__(256) void vq_pack_w(const float* __restrict__ w,
                                                 _Float16* __restrict__ wpack) {
  int t    = blockIdx.x * 256 + threadIdx.x;
  int frag = t >> 9;
  int r    = t & 511;
  int lane = r >> 4;
  int j    = r & 15;
  int nt   = frag >> 1;
  int c    = frag & 1;
  int col  = nt * 16 + (lane & 15);
  int k    = c * 32 + ((lane < 16) ? ((j < 8) ? j : j + 8)
                                   : ((j < 8) ? j + 8 : j + 16));
  wpack[t] = (_Float16)(w[col * VQ_D + k] * 8192.0f);
}

// ---- per-codeword squared norms (f32, unscaled) + histogram zeroing -------
__global__ __launch_bounds__(256) void vq_wnorm(const float* __restrict__ w,
                                                float* __restrict__ wnorm,
                                                int* __restrict__ counts) {
  int k = blockIdx.x * 256 + threadIdx.x;     // 8192 threads
  float s = 0.0f;
  const float* wr = w + (size_t)k * VQ_D;
#pragma unroll 8
  for (int d = 0; d < VQ_D; ++d) { float v = wr[d]; s += v * v; }
  wnorm[k]  = s;
  counts[k] = 0;
}

// ---- fused WMMA GEMM + argmin: one wave per 16-row tile, B via LDS --------
__global__ __launch_bounds__(256) void vq_argmin_wmma(
    const _Float16* __restrict__ xpack, const _Float16* __restrict__ wpack,
    const float* __restrict__ wnorm, int* __restrict__ idx) {
  __shared__ _Float16 btile[TILES_PER_PHASE * 1024];  // 32 KiB

  const int lane = threadIdx.x & 31;
  const int wave = threadIdx.x >> 5;
  const int mt   = blockIdx.x * 8 + wave;     // 128 blocks * 8 waves = 1024

  // A fragments for this row tile stay in VGPRs for the whole K sweep.
  const v16h* xa = (const v16h*)(xpack + (size_t)mt * 1024);
  v16h a0 = xa[lane];        // k-chunk 0 (K=0..31)
  v16h a1 = xa[32 + lane];   // k-chunk 1 (K=32..63)

  float minv[8];
  int   mini[8];
#pragma unroll
  for (int v = 0; v < 8; ++v) { minv[v] = 3.4e38f; mini[v] = 0; }

  const int colLane = lane & 15;
  for (int p = 0; p < NUM_PHASES; ++p) {
    // cooperative stage of 16 column tiles (32 KiB) into LDS
    __syncthreads();
    {
      const uint4* src = (const uint4*)(wpack + (size_t)p * TILES_PER_PHASE * 1024);
      uint4*       dst = (uint4*)btile;
#pragma unroll
      for (int i = 0; i < (TILES_PER_PHASE * 2048 / 16) / 256; ++i)
        dst[i * 256 + threadIdx.x] = src[i * 256 + threadIdx.x];
    }
    __syncthreads();

#pragma unroll 2
    for (int t = 0; t < TILES_PER_PHASE; ++t) {
      const v16h* wb = (const v16h*)(btile + t * 1024);
      v16h b0 = wb[lane];
      v16h b1 = wb[32 + lane];
      v8f acc = {};
      acc = __builtin_amdgcn_wmma_f32_16x16x32_f16(false, a0, false, b0,
                                                   (short)0, acc, false, false);
      acc = __builtin_amdgcn_wmma_f32_16x16x32_f16(false, a1, false, b1,
                                                   (short)0, acc, false, false);
      int   col = (p * TILES_PER_PHASE + t) * 16 + colLane;
      float wn  = wnorm[col];
#pragma unroll
      for (int v = 0; v < 8; ++v) {
        // score = ||w||^2 - 2 * x.w ;  acc = 2^13 * x.w  ->  scale 2^-12
        float s = wn - acc[v] * 0.000244140625f;
        if (s < minv[v]) { minv[v] = s; mini[v] = col; }
      }
    }
  }

  // C layout: VGPR v -> row v (lanes 0-15) / row v+8 (lanes 16-31), col=lane%16.
  // Reduce across the 16 columns held by each half-wave (xor 8,4,2,1 stays
  // within the half). Tie-break: lower index, matching jnp.argmin.
#pragma unroll
  for (int v = 0; v < 8; ++v) {
    float mv = minv[v];
    int   mi = mini[v];
    for (int off = 8; off >= 1; off >>= 1) {
      float ov = __shfl_xor(mv, off, 32);
      int   oi = __shfl_xor(mi, off, 32);
      if (ov < mv || (ov == mv && oi < mi)) { mv = ov; mi = oi; }
    }
    minv[v] = mv; mini[v] = mi;
  }
  if (lane == 0) {
#pragma unroll
    for (int v = 0; v < 8; ++v) idx[mt * 16 + v] = mini[v];
  } else if (lane == 16) {
#pragma unroll
    for (int v = 0; v < 8; ++v) idx[mt * 16 + 8 + v] = mini[v];
  }
}

// ---- gather quantized, straight-through output, loss partials, histogram --
__global__ __launch_bounds__(256) void vq_gather(
    const float* __restrict__ x, const float* __restrict__ w,
    const int* __restrict__ idx, float* __restrict__ out,
    float* __restrict__ partial, int* __restrict__ counts) {
  __shared__ float red[256];
  int t   = blockIdx.x * 256 + threadIdx.x;   // over N*D
  int row = t >> 6;
  int d   = t & 63;
  int k   = idx[row];
  float q  = w[(size_t)k * VQ_D + d];
  float xi = x[t];
  out[t] = xi + (q - xi);                     // straight-through (as reference)
  float diff = q - xi;
  red[threadIdx.x] = diff * diff;
  if (d == 0) atomicAdd(&counts[k], 1);
  __syncthreads();
  for (int s = 128; s > 0; s >>= 1) {
    if (threadIdx.x < s) red[threadIdx.x] += red[threadIdx.x + s];
    __syncthreads();
  }
  if (threadIdx.x == 0) partial[blockIdx.x] = red[0];
}

// ---- deterministic finalize: loss + perplexity ----------------------------
__global__ __launch_bounds__(1024) void vq_finalize(
    const float* __restrict__ partial, const int* __restrict__ counts,
    float* __restrict__ scalars) {
  __shared__ float red[1024];
  float s = 0.0f;
  for (int i = threadIdx.x; i < GATHER_BLOCKS; i += 1024) s += partial[i];
  red[threadIdx.x] = s;
  __syncthreads();
  for (int st = 512; st > 0; st >>= 1) {
    if (threadIdx.x < st) red[threadIdx.x] += red[threadIdx.x + st];
    __syncthreads();
  }
  float loss = red[0] * (1.25f / (float)VQ_NELEM);  // q_loss + 0.25*e_loss
  __syncthreads();
  float e = 0.0f;
  for (int i = threadIdx.x; i < VQ_K; i += 1024) {
    float p = (float)counts[i] * (1.0f / (float)VQ_N);
    e += p * logf(p + 1e-10f);
  }
  red[threadIdx.x] = e;
  __syncthreads();
  for (int st = 512; st > 0; st >>= 1) {
    if (threadIdx.x < st) red[threadIdx.x] += red[threadIdx.x + st];
    __syncthreads();
  }
  if (threadIdx.x == 0) {
    scalars[0] = loss;
    scalars[1] = expf(-red[0]);
  }
}

// ---------------------------------------------------------------------------
extern "C" void kernel_launch(void* const* d_in, const int* in_sizes, int n_in,
                              void* d_out, int out_size, void* d_ws, size_t ws_size,
                              hipStream_t stream) {
  const float* x = (const float*)d_in[0];   // [16,32,32,64] f32
  const float* w = (const float*)d_in[1];   // [8192,64] f32
  float* out = (float*)d_out;               // quantized_st + loss + perplexity

  // workspace layout (bytes)
  char* ws = (char*)d_ws;
  _Float16* xpack   = (_Float16*)(ws);                       // 2 MiB
  _Float16* wpack   = (_Float16*)(ws + (2u << 20));          // 1 MiB
  float*    wnorm   = (float*)   (ws + (3u << 20));          // 32 KiB
  int*      idx     = (int*)     (ws + (3u << 20) + (64u << 10)); // 64 KiB
  int*      counts  = (int*)     (ws + (3u << 20) + (128u << 10)); // 32 KiB
  float*    partial = (float*)   (ws + (3u << 20) + (160u << 10)); // 16 KiB

  vq_pack_x<<<VQ_NELEM / 256, 256, 0, stream>>>(x, xpack);
  vq_pack_w<<<(VQ_K * VQ_D) / 256, 256, 0, stream>>>(w, wpack);
  vq_wnorm<<<VQ_K / 256, 256, 0, stream>>>(w, wnorm, counts);
  vq_argmin_wmma<<<ROW_TILES / 8, 256, 0, stream>>>(xpack, wpack, wnorm, idx);
  vq_gather<<<GATHER_BLOCKS, 256, 0, stream>>>(x, w, idx, out, partial, counts);
  vq_finalize<<<1, 1024, 0, stream>>>(partial, counts, out + VQ_NELEM);
}
